// BertBiLstmCrf_72851235274875
// MI455X (gfx1250) — matrix operand order repaired and checked by reading, not compiled, not run on hardware
//
#include <hip/hip_runtime.h>
#include <hip/hip_bf16.h>

// ---------------------------------------------------------------------------
// BERT BiLSTM-CRF for MI455X (gfx1250, wave32).
//   B=64, T=512, H=768, LH=384, L=12
// Pipeline:
//   K0  f32->bf16 conversions (X, w_ih x2, w_hh x2, fc_w)
//   K1  xproj GEMM: gates_x[dir] = X @ W_ih^T + (b_ih+b_hh)   (bf16 WMMA)
//   K2  persistent recurrent kernel: 12 WGs/dir, grid-sync per step,
//       h_prev staged to LDS via global_load_async_to_lds_b128 (ASYNCcnt),
//       h@W_hh^T via WMMA from LDS, c resident in LDS across all 512 steps
//   K3  FC GEMM -> emissions (WMMA, N padded to 16)
//   K4  Viterbi (one wave per batch, all state in LDS)
// ---------------------------------------------------------------------------

typedef __bf16 bf16_t;
typedef __attribute__((ext_vector_type(16))) __bf16 v16bf;
typedef __attribute__((ext_vector_type(8)))  __bf16 v8bf;
typedef __attribute__((ext_vector_type(8)))  float  v8f;
typedef int v4i_vs __attribute__((vector_size(16)));   // matches builtin's int4

#define BB    64
#define TT    512
#define HDIM  768
#define LHD   384
#define LNUM  12
#define NWG   12          // workgroups per direction in the recurrent kernel
#define NEGV  (-1000.0f)

#if defined(__has_builtin)
#if __has_builtin(__builtin_amdgcn_global_load_async_to_lds_b128) && \
    __has_builtin(__builtin_amdgcn_s_wait_asynccnt)
#define USE_ASYNC_LDS 1
#endif
#endif
#ifndef USE_ASYNC_LDS
#define USE_ASYNC_LDS 0
#endif

#define AS1 __attribute__((address_space(1)))
#define AS3 __attribute__((address_space(3)))

// ---- WMMA fragment loaders (contiguous k order, same for A and B, wave32) --------
// A (16x32, MxK) row-major with leading dim lda:
//   lane l: M = l&15, half = l>>4; elems e -> K = 16*half + e (contig 32B)
__device__ __forceinline__ v16bf load_a_frag(const bf16_t* __restrict__ A,
                                             int lda, int mBase, int k, int lane) {
    const int half = (lane >> 4) & 1;
    const int m    = lane & 15;
    const bf16_t* p = A + (size_t)(mBase + m) * lda + k;
    return *(const v16bf*)(p + 16 * half);
}

// B (32x16, KxN). Weights are stored (N,K) row-major (i.e. W, B = W^T), so
// lane l: N = nBase + (l&15); elems e -> K = k + 16*half + e : one 32B load.
__device__ __forceinline__ v16bf load_b_frag(const bf16_t* __restrict__ W,
                                             int ldb, int nBase, int k, int lane) {
    const int half = (lane >> 4) & 1;
    const int n    = nBase + (lane & 15);
    return *(const v16bf*)(W + (size_t)n * ldb + k + 16 * half);
}

__device__ __forceinline__ float sigf(float x) { return 1.0f / (1.0f + __expf(-x)); }

// 16B async (or fallback sync) copy global -> LDS
__device__ __forceinline__ void copy16_g2lds(const bf16_t* g, bf16_t* lds) {
#if USE_ASYNC_LDS
    __builtin_amdgcn_global_load_async_to_lds_b128(
        (AS1 v4i_vs*)(AS1 void*)(void*)g,
        (AS3 v4i_vs*)(AS3 void*)(void*)lds, 0, 0);
#else
    *(v8bf*)lds = *(const v8bf*)g;
#endif
}

__device__ __forceinline__ void wait_async_copies() {
#if USE_ASYNC_LDS
    __builtin_amdgcn_s_wait_asynccnt(0);
#endif
}

// ---- K0: f32 -> bf16 -------------------------------------------------------
__global__ void f32_to_bf16_kernel(const float* __restrict__ in,
                                   bf16_t* __restrict__ out, int n) {
    int i = blockIdx.x * blockDim.x + threadIdx.x;
    if (i < n) out[i] = (bf16_t)in[i];
}

__global__ void init_counters_kernel(unsigned* __restrict__ c) {
    if (threadIdx.x < 2) c[threadIdx.x] = 0u;
}

// ---- K1: input projection GEMM --------------------------------------------
// C (32768 x 1536) = Xbf (32768 x 768) x W_ih^T, + bias, stored bf16 with
// rows transposed (b*T+t) -> (t*64+b) so the recurrent kernel reads per-t
// contiguous blocks.  block: 256 thr (8 waves); tile M=128, N=64/block.
__global__ __launch_bounds__(256)
void xproj_kernel(const bf16_t* __restrict__ Xbf,
                  const bf16_t* __restrict__ w_ih,     // [2][1536][768]
                  const float*  __restrict__ b_ih_f, const float* __restrict__ b_hh_f,
                  const float*  __restrict__ b_ih_b, const float* __restrict__ b_hh_b,
                  bf16_t* __restrict__ gates_x)        // [2][T*64][1536]
{
    const int dir  = blockIdx.z;
    const int tid  = threadIdx.x;
    const int lane = tid & 31;
    const int wv   = tid >> 5;
    const int half = (lane >> 4) & 1;
    const int mBase  = blockIdx.y * 128 + wv * 16;
    const int nBase0 = blockIdx.x * 64;

    const bf16_t* W  = w_ih + (size_t)dir * 1536 * HDIM;
    const float*  bi = dir ? b_ih_b : b_ih_f;
    const float*  bh = dir ? b_hh_b : b_hh_f;

    v8f acc[4];
    #pragma unroll
    for (int i = 0; i < 4; ++i)
        #pragma unroll
        for (int j = 0; j < 8; ++j) acc[i][j] = 0.0f;

    for (int k = 0; k < HDIM; k += 32) {
        if (k + 64 < HDIM)
            __builtin_prefetch(Xbf + (size_t)(mBase + (lane & 15)) * HDIM + k + 64, 0, 1);
        v16bf a = load_a_frag(Xbf, HDIM, mBase, k, lane);
        #pragma unroll
        for (int i = 0; i < 4; ++i) {
            v16bf b = load_b_frag(W, HDIM, nBase0 + 16 * i, k, lane);
            acc[i] = __builtin_amdgcn_wmma_f32_16x16x32_bf16(
                         false, a, false, b, (short)0, acc[i], false, false);
        }
    }

    bf16_t* gdst = gates_x + (size_t)dir * TT * BB * 1536;
    #pragma unroll
    for (int i = 0; i < 4; ++i) {
        const int col = nBase0 + 16 * i + (lane & 15);
        const float bias = bi[col] + bh[col];
        #pragma unroll
        for (int j = 0; j < 8; ++j) {
            const int r  = mBase + j + 8 * half;          // r = b*512 + t
            const int tr = ((r & 511) << 6) + (r >> 9);   // t*64 + b
            gdst[(size_t)tr * 1536 + col] = (bf16_t)(acc[i][j] + bias);
        }
    }
}

// ---- grid barrier for the persistent recurrent kernel ----------------------
__device__ __forceinline__ void grid_arrive_wait(unsigned* cnt, unsigned target) {
    __threadfence();
    __syncthreads();
    if (threadIdx.x == 0) {
        __hip_atomic_fetch_add(cnt, 1u, __ATOMIC_ACQ_REL, __HIP_MEMORY_SCOPE_AGENT);
        while (__hip_atomic_load(cnt, __ATOMIC_ACQUIRE, __HIP_MEMORY_SCOPE_AGENT) < target) {
            __builtin_amdgcn_s_sleep(2);
        }
    }
    __syncthreads();
    __threadfence();
}

// ---- K2: persistent recurrent kernel ---------------------------------------
// grid = (12 WGs, 2 dirs); block = 256 (8 waves).  Each WG owns h columns
// [wg*32, wg*32+32) and the 4 matching gate column groups (128 gate cols).
// Per step:
//   1. async-stage h_prev (64x384 bf16, 48KB) global -> LDS     (ASYNCcnt)
//   2. (64 x 384) x (384 x 128) WMMA GEMM, A-fragments from LDS
//   3. gate partials staged to LDS (same region, disjoint lifetime)
//   4. pointwise LSTM cell, c resident in LDS, vectorized 16B accesses
//   5. grid sync via L2 atomic counter
__global__ __launch_bounds__(256)
void lstm_recurrent_kernel(const bf16_t* __restrict__ gates_x,  // [2][T*64][1536]
                           const bf16_t* __restrict__ w_hh,     // [2][1536][384]
                           const float*  __restrict__ h0,       // (2,64,384)
                           const float*  __restrict__ c0,       // (2,64,384)
                           bf16_t* __restrict__ h_buf,          // [2][2][64*384]
                           bf16_t* __restrict__ lstm_out,       // [B*T][768]
                           unsigned* __restrict__ counters)     // [2]
{
    const int wg   = blockIdx.x;       // 0..11
    const int dir  = blockIdx.y;       // 0..1
    const int tid  = threadIdx.x;
    const int lane = tid & 31;
    const int wv   = tid >> 5;
    const int half = (lane >> 4) & 1;

    // 48KB shared region: h staging (bf16 64x384) and gate staging
    // (f32 64x128 = 32KB) have disjoint lifetimes within a step.
    __shared__ __align__(32) unsigned char smem[BB * LHD * 2];
    __shared__ float c_slice[BB][32];    // persistent cell state (8 KB)
    bf16_t* h_lds = (bf16_t*)smem;
    float*  g_lds = (float*)smem;        // [BB][128]

    const bf16_t* gx  = gates_x + (size_t)dir * TT * BB * 1536;
    const bf16_t* whh = w_hh    + (size_t)dir * 1536 * LHD;
    unsigned*     cnt = counters + dir;

    // init cell state and h parity-buffer 0 (each WG fills its own columns)
    bf16_t* hb0 = h_buf + ((size_t)dir * 2 + 0) * (BB * LHD);
    for (int p = tid; p < BB * 32; p += 256) {
        const int b = p >> 5, jj = p & 31;
        const size_t src = ((size_t)dir * BB + b) * LHD + wg * 32 + jj;
        c_slice[b][jj] = c0[src];
        hb0[(size_t)b * LHD + wg * 32 + jj] = (bf16_t)h0[src];
    }
    unsigned phase = 1;
    grid_arrive_wait(cnt, phase * NWG);

    const int mBase = (wv & 3) * 16;          // 4 M-tiles of 16 (batch)
    const int nb0   = (wv >> 2) * 4;          // 8 N-tiles of 16, 4 per wave-half
    const int pb    = tid >> 2;               // pointwise: batch row 0..63
    const int pj0   = (tid & 3) * 8;          // pointwise: 8 contiguous cols

    for (int s = 0; s < TT; ++s) {
        const int t = dir ? (TT - 1 - s) : s;
        const bf16_t* hprev = h_buf + ((size_t)dir * 2 + (s & 1)) * (BB * LHD);
        bf16_t*       hnext = h_buf + ((size_t)dir * 2 + ((s + 1) & 1)) * (BB * LHD);

        // 1. stage h_prev into LDS: 3072 x 16B chunks, 12 per thread
        #pragma unroll
        for (int c = 0; c < 12; ++c) {
            const int idx = (c * 256 + tid) * 8;    // bf16 elements
            copy16_g2lds(hprev + idx, h_lds + idx);
        }
        wait_async_copies();
        __syncthreads();

        // 2. WMMA GEMM, A from LDS
        v8f acc[4];
        #pragma unroll
        for (int i = 0; i < 4; ++i)
            #pragma unroll
            for (int j = 0; j < 8; ++j) acc[i][j] = 0.0f;

        for (int k = 0; k < LHD; k += 32) {
            v16bf a = load_a_frag(h_lds, LHD, mBase, k, lane);
            #pragma unroll
            for (int i = 0; i < 4; ++i) {
                const int cl0  = (nb0 + i) * 16;                 // local col base
                const int gate = cl0 >> 5;
                const int colg = gate * LHD + wg * 32 + (cl0 & 31);
                v16bf b = load_b_frag(whh, LHD, colg, k, lane);
                acc[i] = __builtin_amdgcn_wmma_f32_16x16x32_bf16(
                             false, a, false, b, (short)0, acc[i], false, false);
            }
        }
        __syncthreads();   // all h_lds reads done before overwriting as g_lds

        // 3. stage gate partials to LDS
        #pragma unroll
        for (int i = 0; i < 4; ++i) {
            const int cl = (nb0 + i) * 16 + (lane & 15);
            #pragma unroll
            for (int j = 0; j < 8; ++j)
                g_lds[(mBase + j + 8 * half) * 128 + cl] = acc[i][j];
        }
        __syncthreads();

        // 4. pointwise LSTM cell: thread -> (batch pb, cols pj0..pj0+7)
        {
            const bf16_t* gxr = gx + ((size_t)t * BB + pb) * 1536 + wg * 32 + pj0;
            v8bf gi = *(const v8bf*)(gxr + 0 * LHD);
            v8bf gf = *(const v8bf*)(gxr + 1 * LHD);
            v8bf gg = *(const v8bf*)(gxr + 2 * LHD);
            v8bf go = *(const v8bf*)(gxr + 3 * LHD);
            v8bf hv;
            #pragma unroll
            for (int e = 0; e < 8; ++e) {
                const float ig = g_lds[pb * 128 +  0 + pj0 + e] + (float)gi[e];
                const float fg = g_lds[pb * 128 + 32 + pj0 + e] + (float)gf[e];
                const float gv = g_lds[pb * 128 + 64 + pj0 + e] + (float)gg[e];
                const float og = g_lds[pb * 128 + 96 + pj0 + e] + (float)go[e];
                const float c  = sigf(fg) * c_slice[pb][pj0 + e] + sigf(ig) * tanhf(gv);
                const float h  = sigf(og) * tanhf(c);
                c_slice[pb][pj0 + e] = c;
                hv[e] = (bf16_t)h;
            }
            *(v8bf*)(hnext + (size_t)pb * LHD + wg * 32 + pj0) = hv;
            *(v8bf*)(lstm_out + ((size_t)pb * TT + t) * (2 * LHD)
                              + dir * LHD + wg * 32 + pj0) = hv;
        }

        ++phase;
        grid_arrive_wait(cnt, phase * NWG);
    }
}

// ---- K3: FC to emissions ----------------------------------------------------
__global__ __launch_bounds__(256)
void fc_kernel(const bf16_t* __restrict__ lstm_out,  // [32768][768]
               const bf16_t* __restrict__ fcw,       // [12][768]
               const float*  __restrict__ fcb,       // [12]
               float* __restrict__ emis)             // [32768][12]
{
    const int tid  = threadIdx.x;
    const int lane = tid & 31;
    const int wv   = tid >> 5;
    const int half = (lane >> 4) & 1;
    const int mBase = (blockIdx.x * 8 + wv) * 16;
    const int n     = lane & 15;

    v16bf zb;
    #pragma unroll
    for (int e = 0; e < 16; ++e) zb[e] = (bf16_t)0.0f;

    v8f acc;
    #pragma unroll
    for (int j = 0; j < 8; ++j) acc[j] = 0.0f;

    for (int k = 0; k < HDIM; k += 32) {
        v16bf a = load_a_frag(lstm_out, HDIM, mBase, k, lane);
        v16bf b = zb;
        if (n < LNUM)
            b = *(const v16bf*)(fcw + (size_t)n * HDIM + k + 16 * half);
        acc = __builtin_amdgcn_wmma_f32_16x16x32_bf16(
                  false, a, false, b, (short)0, acc, false, false);
    }
    if (n < LNUM) {
        const float bias = fcb[n];
        #pragma unroll
        for (int j = 0; j < 8; ++j)
            emis[(size_t)(mBase + j + 8 * half) * LNUM + n] = acc[j] + bias;
    }
}

// ---- K4: Viterbi ------------------------------------------------------------
__global__ __launch_bounds__(32)
void viterbi_kernel(const float* __restrict__ emis,   // [(b*T+t)][12]
                    const float* __restrict__ trans,  // [12][12]
                    const int*   __restrict__ start_idx,
                    float* __restrict__ out_score,    // [64]
                    float* __restrict__ out_path)     // [64][512]
{
    const int b    = blockIdx.x;
    const int lane = threadIdx.x;
    __shared__ float tr[LNUM * LNUM];
    __shared__ float fv[LNUM];
    __shared__ unsigned char bp[TT][LNUM];

    for (int p = lane; p < LNUM * LNUM; p += 32) tr[p] = trans[p];
    if (lane < LNUM) fv[lane] = (lane == start_idx[0]) ? 0.0f : NEGV;
    __syncthreads();

    const float* erow = emis + (size_t)b * TT * LNUM;
    for (int t = 1; t < TT; ++t) {
        float best = -3.4e38f;
        int   arg  = 0;
        if (lane < LNUM) {
            #pragma unroll
            for (int p = 0; p < LNUM; ++p) {
                const float s = tr[lane * LNUM + p] + fv[p];
                if (s > best) { best = s; arg = p; }
            }
            best += erow[t * LNUM + lane];
        }
        __syncthreads();
        if (lane < LNUM) { fv[lane] = best; bp[t][lane] = (unsigned char)arg; }
        __syncthreads();
    }

    if (lane == 0) {
        float bs = fv[0];
        int last = 0;
        for (int l = 1; l < LNUM; ++l)
            if (fv[l] > bs) { bs = fv[l]; last = l; }
        out_score[b] = bs;
        int lab = last;
        for (int t = TT - 1; t >= 1; --t) {
            out_path[(size_t)b * TT + t] = (float)lab;
            lab = bp[t][lab];
        }
        out_path[(size_t)b * TT + 0] = (float)lab;
    }
}

// ---------------------------------------------------------------------------
extern "C" void kernel_launch(void* const* d_in, const int* in_sizes, int n_in,
                              void* d_out, int out_size, void* d_ws, size_t ws_size,
                              hipStream_t stream) {
    (void)in_sizes; (void)n_in; (void)out_size; (void)ws_size;

    const float* hs      = (const float*)d_in[0];
    const float* h0      = (const float*)d_in[1];
    const float* c0      = (const float*)d_in[2];
    const float* w_ih_f  = (const float*)d_in[3];
    const float* w_hh_f  = (const float*)d_in[4];
    const float* b_ih_f  = (const float*)d_in[5];
    const float* b_hh_f  = (const float*)d_in[6];
    const float* w_ih_b  = (const float*)d_in[7];
    const float* w_hh_b  = (const float*)d_in[8];
    const float* b_ih_b  = (const float*)d_in[9];
    const float* b_hh_b  = (const float*)d_in[10];
    const float* fc_w    = (const float*)d_in[11];
    const float* fc_b    = (const float*)d_in[12];
    const float* transit = (const float*)d_in[13];
    const int*   start_i = (const int*)d_in[14];

    float* out_score = (float*)d_out;              // [64]
    float* out_path  = (float*)d_out + BB;         // [64][512]

    // ---- workspace carving (≈310 MB) ----
    char*  ws  = (char*)d_ws;
    size_t off = 0;
    auto carve = [&](size_t bytes) -> void* {
        off = (off + 255) & ~(size_t)255;
        void* p = ws + off;
        off += bytes;
        return p;
    };
    const size_t nX = (size_t)BB * TT * HDIM;            // 25,165,824
    bf16_t*  Xbf      = (bf16_t*)carve(nX * 2);
    bf16_t*  wih_bf   = (bf16_t*)carve((size_t)2 * 1536 * HDIM * 2);
    bf16_t*  whh_bf   = (bf16_t*)carve((size_t)2 * 1536 * LHD * 2);
    bf16_t*  fcw_bf   = (bf16_t*)carve((size_t)LNUM * HDIM * 2);
    bf16_t*  gates_x  = (bf16_t*)carve((size_t)2 * TT * BB * 1536 * 2);
    bf16_t*  h_buf    = (bf16_t*)carve((size_t)2 * 2 * BB * LHD * 2);
    bf16_t*  lstm_out = (bf16_t*)carve((size_t)BB * TT * 2 * LHD * 2);
    float*   emis     = (float*)carve((size_t)BB * TT * LNUM * 4);
    unsigned* counters = (unsigned*)carve(64);

    // ---- K0: conversions ----
    auto cvt = [&](const float* in, bf16_t* out, size_t n) {
        const int blk = 256;
        const int grd = (int)((n + blk - 1) / blk);
        f32_to_bf16_kernel<<<grd, blk, 0, stream>>>(in, out, (int)n);
    };
    cvt(hs,     Xbf,                         nX);
    cvt(w_ih_f, wih_bf,                      (size_t)1536 * HDIM);
    cvt(w_ih_b, wih_bf + (size_t)1536*HDIM,  (size_t)1536 * HDIM);
    cvt(w_hh_f, whh_bf,                      (size_t)1536 * LHD);
    cvt(w_hh_b, whh_bf + (size_t)1536*LHD,   (size_t)1536 * LHD);
    cvt(fc_w,   fcw_bf,                      (size_t)LNUM * HDIM);
    init_counters_kernel<<<1, 32, 0, stream>>>(counters);

    // ---- K1: input projection ----
    {
        dim3 grid(1536 / 64, (BB * TT) / 128, 2);
        xproj_kernel<<<grid, 256, 0, stream>>>(Xbf, wih_bf,
                                               b_ih_f, b_hh_f, b_ih_b, b_hh_b,
                                               gates_x);
    }

    // ---- K2: persistent bidirectional recurrence ----
    {
        dim3 grid(NWG, 2);
        lstm_recurrent_kernel<<<grid, 256, 0, stream>>>(gates_x, whh_bf, h0, c0,
                                                        h_buf, lstm_out, counters);
    }

    // ---- K3: FC ----
    fc_kernel<<<(BB * TT) / 128, 256, 0, stream>>>(lstm_out, fcw_bf, fc_b, emis);

    // ---- K4: Viterbi ----
    viterbi_kernel<<<BB, 32, 0, stream>>>(emis, transit, start_i,
                                          out_score, out_path);
}
